// SpaceTimeAtten_57844619542888
// MI455X (gfx1250) — matrix operand order repaired and checked by reading, not compiled, not run on hardware
//
#include <hip/hip_runtime.h>
#include <cstdint>
#include <cstddef>

// ---------------------------------------------------------------------------
// CDNA5 (gfx1250) wave32 WMMA implementation of SpaceTimeAtten.
// All matrix math runs on v_wmma_f32_16x16x32_bf16 (f32 accumulate).
// ---------------------------------------------------------------------------

typedef __bf16 bf16_t;
typedef bf16_t v16bf __attribute__((ext_vector_type(16)));
typedef float  v8f   __attribute__((ext_vector_type(8)));

constexpr int   NB     = 2;
constexpr int   C      = 512;
constexpr int   THW    = 6272;                 // 8*28*28
constexpr float BN_EPS = 1e-5f;
constexpr float NEG_BIG = -3.4e38f;

// float -> bf16, round-to-nearest-even (bit-exact, no reliance on cast support)
__device__ __forceinline__ bf16_t f2bf(float f) {
  union { float f; uint32_t u; } v; v.f = f;
  uint32_t r = (v.u + 0x7FFFu + ((v.u >> 16) & 1u)) >> 16;
  union { uint16_t s; bf16_t b; } o; o.s = (uint16_t)r;
  return o.b;
}

// A-fragment (M=16 x K=32 bf16), source stored row-major [M][ld].
// ISA layout: lanes 0-15 -> row=lane, K halves {0..7, 16..23};
//             lanes 16-31 -> row=lane-16, K halves {8..15, 24..31}.
__device__ __forceinline__ v16bf load_afrag(const bf16_t* base, int ld, int lane) {
  int row  = lane & 15;
  int koff = (lane & 16) ? 8 : 0;
  const bf16_t* p = base + row * ld + koff;
  union { v16bf v; uint4 q[2]; } u;
  u.q[0] = *(const uint4*)(p);        // K = koff .. koff+7
  u.q[1] = *(const uint4*)(p + 16);   // K = koff+16 .. koff+23
  return u.v;
}

// B-fragment (K=32 x N=16 bf16), source stored as [N][ld] (contiguous K per col).
// ISA layout: lanes 0-15 -> col=lane, K halves {0..7, 16..23};
//             lanes 16-31 -> col=lane-16, K halves {8..15, 24..31}.
__device__ __forceinline__ v16bf load_bfrag(const bf16_t* base, int ld, int lane) {
  int col  = lane & 15;
  int koff = (lane & 16) ? 8 : 0;
  const bf16_t* p = base + col * ld + koff;
  union { v16bf v; uint4 q[2]; } u;
  u.q[0] = *(const uint4*)(p);
  u.q[1] = *(const uint4*)(p + 16);
  return u.v;
}

__device__ __forceinline__ v8f wmma_bf16(v16bf a, v16bf b, v8f c) {
  // (neg_a, A, neg_b, B, c_mod, C, reuse_a, reuse_b)
  return __builtin_amdgcn_wmma_f32_16x16x32_bf16(false, a, false, b, (short)0, c,
                                                 false, false);
}

__device__ __forceinline__ v8f zero8() {
  v8f z; for (int r = 0; r < 8; ++r) z[r] = 0.f; return z;
}

// ---------------------------------------------------------------------------
// Phase A: projection GEMM  Y[n,o,p] = sum_c W[o,c] * X[n,c,p] + bias[o]
// MODE 0: bf16 out, transposed (p-major)   -> used by ph_x, pg
// MODE 1: bf16 out, natural (c-major)      -> used by ph_m
// MODE 2: f32  out, natural                -> used by pm, wz
// Block tile 64(M) x 128(P), 8 waves in 2x4, wave tile 32x32 (4 WMMA tiles).
// ---------------------------------------------------------------------------
template <int MODE>
__global__ __launch_bounds__(256) void proj_gemm(const float* __restrict__ X,
                                                 const float* __restrict__ W,
                                                 const float* __restrict__ bias,
                                                 void* __restrict__ Yv) {
  __shared__ bf16_t As[64 * 32];    // [m][k] row-major
  __shared__ bf16_t Bs[128 * 32];   // [p][k] (transposed from X)

  const int pt = blockIdx.x;        // 0..48
  const int mt = blockIdx.y;        // 0..7
  const int n  = blockIdx.z;        // 0..1
  const int tid = threadIdx.x;
  const int lane = tid & 31, wave = tid >> 5;
  const int wm = (wave >> 2) * 32;  // wave row offset within 64
  const int wp = (wave & 3) * 32;   // wave col offset within 128
  const int m0 = mt * 64;
  const int p0 = pt * 128;
  const float* Xn = X + (size_t)n * C * THW;

  v8f acc[2][2];
  for (int i = 0; i < 2; ++i)
    for (int j = 0; j < 2; ++j) acc[i][j] = zero8();

  for (int k0 = 0; k0 < C; k0 += 32) {
    for (int i = tid; i < 64 * 32; i += 256) {
      int r = i >> 5, c = i & 31;
      As[i] = f2bf(W[(size_t)(m0 + r) * C + (k0 + c)]);
    }
    for (int i = tid; i < 32 * 128; i += 256) {
      int k = i >> 7, p = i & 127;
      Bs[p * 32 + k] = f2bf(Xn[(size_t)(k0 + k) * THW + (p0 + p)]);
    }
    __syncthreads();

    v16bf a0 = load_afrag(As + (size_t)wm * 32, 32, lane);
    v16bf a1 = load_afrag(As + (size_t)(wm + 16) * 32, 32, lane);
    v16bf b0 = load_bfrag(Bs + (size_t)wp * 32, 32, lane);
    v16bf b1 = load_bfrag(Bs + (size_t)(wp + 16) * 32, 32, lane);
    acc[0][0] = wmma_bf16(a0, b0, acc[0][0]);
    acc[0][1] = wmma_bf16(a0, b1, acc[0][1]);
    acc[1][0] = wmma_bf16(a1, b0, acc[1][0]);
    acc[1][1] = wmma_bf16(a1, b1, acc[1][1]);
    __syncthreads();
  }

  // D layout: lanes 0-15 -> rows r, col=lane; lanes 16-31 -> rows 8+r, col=lane-16
  const int colL = lane & 15;
  const int rOff = (lane & 16) ? 8 : 0;
  for (int ti = 0; ti < 2; ++ti)
    for (int tj = 0; tj < 2; ++tj)
      for (int r = 0; r < 8; ++r) {
        int m = m0 + wm + ti * 16 + rOff + r;
        int p = p0 + wp + tj * 16 + colL;
        float v = acc[ti][tj][r] + bias[m];
        if (MODE == 0)
          ((bf16_t*)Yv)[((size_t)n * THW + p) * C + m] = f2bf(v);
        else if (MODE == 1)
          ((bf16_t*)Yv)[((size_t)n * C + m) * THW + p] = f2bf(v);
        else
          ((float*)Yv)[((size_t)n * C + m) * THW + p] = v;
      }
}

// ---------------------------------------------------------------------------
// Phase B1: per-row softmax stats of energy[t,s] = sum_c phxT[t,c]*pgT[s,c].
// One block per (16-row t-tile, n). 8 waves stride over s in 16-col tiles,
// each keeping an online (max, sum-exp); combined across waves via LDS.
// ---------------------------------------------------------------------------
__global__ __launch_bounds__(256) void attn_stats(const bf16_t* __restrict__ phxT,
                                                  const bf16_t* __restrict__ pgT,
                                                  float* __restrict__ rowMax,
                                                  float* __restrict__ rowSum) {
  __shared__ bf16_t Axt[16 * 512];
  __shared__ float redM[16][8];
  __shared__ float redL[16][8];

  const int tt = blockIdx.x, n = blockIdx.y;
  const int tid = threadIdx.x, lane = tid & 31, wave = tid >> 5;
  const int t0 = tt * 16;

  {  // 16KB contiguous copy of the t-block of phxT (p-major layout)
    const uint4* src = (const uint4*)(phxT + ((size_t)n * THW + t0) * C);
    uint4* dst = (uint4*)Axt;
    for (int i = tid; i < (16 * 512) / 8; i += 256) dst[i] = src[i];
  }
  __syncthreads();

  float m[8], l[8];
  for (int r = 0; r < 8; ++r) { m[r] = NEG_BIG; l[r] = 0.f; }

  const bf16_t* pgn = pgT + (size_t)n * THW * C;
  for (int s = wave * 16; s < THW; s += 128) {
    const bf16_t* bb = pgn + (size_t)s * C;
    __builtin_prefetch(bb + 128 * C, 0, 0);   // global_prefetch next chunk
    v8f e = zero8();
    for (int c0 = 0; c0 < C; c0 += 32) {
      v16bf a = load_afrag(Axt + c0, C, lane);
      v16bf b = load_bfrag(bb + c0, C, lane);
      e = wmma_bf16(a, b, e);
    }
    // per-row online max / sum-exp (reduce across the 16 columns in the half)
    for (int r = 0; r < 8; ++r) {
      float v = e[r];
      float mx = v;
      for (int off = 1; off < 16; off <<= 1)
        mx = fmaxf(mx, __shfl_xor(mx, off, 32));
      float nm = fmaxf(m[r], mx);
      float p = __expf(v - nm);
      float ss = p;
      for (int off = 1; off < 16; off <<= 1)
        ss += __shfl_xor(ss, off, 32);
      l[r] = l[r] * __expf(m[r] - nm) + ss;
      m[r] = nm;
    }
  }

  if ((lane & 15) == 0) {
    int rb = (lane & 16) ? 8 : 0;
    for (int r = 0; r < 8; ++r) { redM[rb + r][wave] = m[r]; redL[rb + r][wave] = l[r]; }
  }
  __syncthreads();
  if (tid < 16) {
    float M = NEG_BIG;
    for (int w = 0; w < 8; ++w) M = fmaxf(M, redM[tid][w]);
    float L = 0.f;
    for (int w = 0; w < 8; ++w) L += redL[tid][w] * __expf(redM[tid][w] - M);
    rowMax[(size_t)n * THW + t0 + tid] = M;
    rowSum[(size_t)n * THW + t0 + tid] = L;
  }
}

// ---------------------------------------------------------------------------
// Phase B2: fused P = softmax(energy) and ME[t,c] = sum_s P[t,s]*ph_m[c,s].
// Per 128-s chunk: wave w computes its 16-wide E sub-tile, exponentiates with
// precomputed stats, stages P(bf16) in LDS; then each wave accumulates its own
// 64 output channels with WMMA (B-fragments straight from L2-resident ph_m).
// ---------------------------------------------------------------------------
__global__ __launch_bounds__(256) void attn_apply(const bf16_t* __restrict__ phxT,
                                                  const bf16_t* __restrict__ pgT,
                                                  const bf16_t* __restrict__ phm,
                                                  const float* __restrict__ rowMax,
                                                  const float* __restrict__ rowSum,
                                                  float* __restrict__ me) {
  __shared__ bf16_t Axt[16 * 512];
  __shared__ bf16_t Pt[16 * 128];
  __shared__ float sM[16], sL[16];

  const int tt = blockIdx.x, n = blockIdx.y;
  const int tid = threadIdx.x, lane = tid & 31, wave = tid >> 5;
  const int t0 = tt * 16;

  {
    const uint4* src = (const uint4*)(phxT + ((size_t)n * THW + t0) * C);
    uint4* dst = (uint4*)Axt;
    for (int i = tid; i < (16 * 512) / 8; i += 256) dst[i] = src[i];
  }
  if (tid < 16) {
    sM[tid] = rowMax[(size_t)n * THW + t0 + tid];
    sL[tid] = rowSum[(size_t)n * THW + t0 + tid];
  }
  __syncthreads();

  const int rOff = (lane & 16) ? 8 : 0;
  float rM[8], rInv[8];
  for (int r = 0; r < 8; ++r) {
    rM[r] = sM[rOff + r];
    float L = sL[rOff + r];
    rInv[r] = (L > 0.f) ? 1.f / L : 0.f;
  }

  const int c0w = wave * 64;  // this wave owns 64 output channels
  v8f O[4];
  for (int ct = 0; ct < 4; ++ct) O[ct] = zero8();

  const bf16_t* pgn = pgT + (size_t)n * THW * C;
  for (int s0 = 0; s0 < THW; s0 += 128) {
    const int sw = s0 + wave * 16;
    const bf16_t* bb = pgn + (size_t)sw * C;
    __builtin_prefetch(bb + 128 * C, 0, 0);
    v8f e = zero8();
    for (int c0 = 0; c0 < C; c0 += 32) {
      v16bf a = load_afrag(Axt + c0, C, lane);
      v16bf b = load_bfrag(bb + c0, C, lane);
      e = wmma_bf16(a, b, e);
    }
    for (int r = 0; r < 8; ++r) {
      float p = __expf(e[r] - rM[r]) * rInv[r];
      Pt[(rOff + r) * 128 + wave * 16 + (lane & 15)] = f2bf(p);
    }
    __syncthreads();

    for (int st = 0; st < 4; ++st) {
      v16bf a = load_afrag(Pt + st * 32, 128, lane);
      for (int ct = 0; ct < 4; ++ct) {
        const bf16_t* bp =
            phm + ((size_t)n * C + c0w + ct * 16) * THW + (s0 + st * 32);
        v16bf b = load_bfrag(bp, THW, lane);
        O[ct] = wmma_bf16(a, b, O[ct]);
      }
    }
    __syncthreads();
  }

  for (int ct = 0; ct < 4; ++ct)
    for (int r = 0; r < 8; ++r) {
      int c = c0w + ct * 16 + (lane & 15);
      int t = t0 + rOff + r;
      me[((size_t)n * C + c) * THW + t] = O[ct][r];
    }
}

// ---------------------------------------------------------------------------
// BatchNorm batch stats over (N, THW) per channel.
// ---------------------------------------------------------------------------
__global__ __launch_bounds__(256) void bn_stats(const float* __restrict__ wz,
                                                float* __restrict__ mu,
                                                float* __restrict__ rs) {
  __shared__ float s1[256], s2[256];
  const int c = blockIdx.x, tid = threadIdx.x;
  float a = 0.f, b = 0.f;
  for (int n = 0; n < NB; ++n) {
    const float* p = wz + ((size_t)n * C + c) * THW;
    for (int i = tid; i < THW; i += 256) { float v = p[i]; a += v; b += v * v; }
  }
  s1[tid] = a; s2[tid] = b; __syncthreads();
  for (int o = 128; o > 0; o >>= 1) {
    if (tid < o) { s1[tid] += s1[tid + o]; s2[tid] += s2[tid + o]; }
    __syncthreads();
  }
  if (tid == 0) {
    float inv = 1.f / (float)(NB * THW);
    float mean = s1[0] * inv;
    float var = s2[0] * inv - mean * mean;
    mu[c] = mean;
    rs[c] = rsqrtf(var + BN_EPS);
  }
}

// ---------------------------------------------------------------------------
// Per-(n,c) softmax stats of mask_energy over THW.
// ---------------------------------------------------------------------------
__global__ __launch_bounds__(256) void ch_stats(const float* __restrict__ me,
                                                float* __restrict__ cmax,
                                                float* __restrict__ csum) {
  __shared__ float sm[256];
  const int row = blockIdx.x;  // n*C + c
  const float* p = me + (size_t)row * THW;
  const int tid = threadIdx.x;
  float mx = NEG_BIG;
  for (int i = tid; i < THW; i += 256) mx = fmaxf(mx, p[i]);
  sm[tid] = mx; __syncthreads();
  for (int o = 128; o > 0; o >>= 1) {
    if (tid < o) sm[tid] = fmaxf(sm[tid], sm[tid + o]);
    __syncthreads();
  }
  const float M = sm[0];
  __syncthreads();
  float s = 0.f;
  for (int i = tid; i < THW; i += 256) s += __expf(p[i] - M);
  sm[tid] = s; __syncthreads();
  for (int o = 128; o > 0; o >>= 1) {
    if (tid < o) sm[tid] += sm[tid + o];
    __syncthreads();
  }
  if (tid == 0) { cmax[row] = M; csum[row] = sm[0]; }
}

// ---------------------------------------------------------------------------
// Fused epilogue: out = gamma * pm * softmax(me) + BN(wz).
// ---------------------------------------------------------------------------
__global__ __launch_bounds__(256) void final_fuse(const float* __restrict__ pm,
                                                  const float* __restrict__ wz,
                                                  const float* __restrict__ me,
                                                  const float* __restrict__ cmax,
                                                  const float* __restrict__ csum,
                                                  const float* __restrict__ mu,
                                                  const float* __restrict__ rs,
                                                  const float* __restrict__ bnw,
                                                  const float* __restrict__ bnb,
                                                  const float* __restrict__ gamma,
                                                  float* __restrict__ out) {
  size_t i = (size_t)blockIdx.x * 256 + threadIdx.x;
  if (i >= (size_t)NB * C * THW) return;
  int row = (int)(i / THW);  // n*C + c
  int c = row % C;
  float att = __expf(me[i] - cmax[row]) / csum[row];
  float bn = (wz[i] - mu[c]) * rs[c] * bnw[c] + bnb[c];
  out[i] = gamma[0] * pm[i] * att + bn;
}

// ---------------------------------------------------------------------------
// Host launcher.
// ---------------------------------------------------------------------------
extern "C" void kernel_launch(void* const* d_in, const int* in_sizes, int n_in,
                              void* d_out, int out_size, void* d_ws, size_t ws_size,
                              hipStream_t stream) {
  const float* x    = (const float*)d_in[0];
  const float* mask = (const float*)d_in[1];
  const float* Wh   = (const float*)d_in[2];
  const float* bh   = (const float*)d_in[3];
  const float* Wg   = (const float*)d_in[4];
  const float* bg   = (const float*)d_in[5];
  const float* Wm   = (const float*)d_in[6];
  const float* bm   = (const float*)d_in[7];
  const float* Wz   = (const float*)d_in[8];
  const float* bz   = (const float*)d_in[9];
  const float* bnw  = (const float*)d_in[10];
  const float* bnb  = (const float*)d_in[11];
  const float* gam  = (const float*)d_in[12];
  float* out = (float*)d_out;

  const size_t nE   = (size_t)NB * C * THW;        // 6,422,528 per-tensor elems
  char* ws = (char*)d_ws;
  size_t off = 0;
  bf16_t* phxT = (bf16_t*)(ws + off); off += nE * sizeof(bf16_t);   // (n,p,c)
  bf16_t* pgT  = (bf16_t*)(ws + off); off += nE * sizeof(bf16_t);   // (n,p,c)
  bf16_t* phm  = (bf16_t*)(ws + off); off += nE * sizeof(bf16_t);   // (n,c,p)
  float*  pm   = (float*) (ws + off); off += nE * sizeof(float);
  float*  wz   = (float*) (ws + off); off += nE * sizeof(float);
  float*  me   = (float*) (ws + off); off += nE * sizeof(float);
  float*  rMax = (float*) (ws + off); off += (size_t)NB * THW * sizeof(float);
  float*  rSum = (float*) (ws + off); off += (size_t)NB * THW * sizeof(float);
  float*  cMax = (float*) (ws + off); off += (size_t)NB * C * sizeof(float);
  float*  cSum = (float*) (ws + off); off += (size_t)NB * C * sizeof(float);
  float*  mu   = (float*) (ws + off); off += (size_t)C * sizeof(float);
  float*  rs   = (float*) (ws + off); off += (size_t)C * sizeof(float);
  (void)ws_size; (void)n_in; (void)in_sizes; (void)out_size;

  dim3 gGemm(THW / 128, C / 64, NB);   // 49 x 8 x 2
  proj_gemm<0><<<gGemm, 256, 0, stream>>>(x,    Wh, bh, (void*)phxT);
  proj_gemm<0><<<gGemm, 256, 0, stream>>>(x,    Wg, bg, (void*)pgT);
  proj_gemm<1><<<gGemm, 256, 0, stream>>>(mask, Wh, bh, (void*)phm);
  proj_gemm<2><<<gGemm, 256, 0, stream>>>(x,    Wm, bm, (void*)pm);
  proj_gemm<2><<<gGemm, 256, 0, stream>>>(x,    Wz, bz, (void*)wz);

  dim3 gAttn(THW / 16, NB);            // 392 x 2
  attn_stats<<<gAttn, 256, 0, stream>>>(phxT, pgT, rMax, rSum);
  attn_apply<<<gAttn, 256, 0, stream>>>(phxT, pgT, phm, rMax, rSum, me);

  bn_stats<<<C, 256, 0, stream>>>(wz, mu, rs);
  ch_stats<<<NB * C, 256, 0, stream>>>(me, cMax, cSum);

  int nBlocks = (int)((nE + 255) / 256);
  final_fuse<<<nBlocks, 256, 0, stream>>>(pm, wz, me, cMax, cSum, mu, rs,
                                          bnw, bnb, gam, out);
}
